// MultiHeadAttention_2138893714149
// MI455X (gfx1250) — matrix operand order repaired and checked by reading, not compiled, not run on hardware
//
#include <hip/hip_runtime.h>
#include <hip/hip_bf16.h>

#define B_  2
#define S_  2048
#define D_  2048
#define H_  16
#define HD_ 128

typedef __attribute__((ext_vector_type(16))) __bf16 bf16x16;
typedef __attribute__((ext_vector_type(8)))  float  f32x8;
typedef __attribute__((ext_vector_type(4)))  int    v4i;

union ABFrag { unsigned int u[8]; bf16x16 v; };

__device__ __forceinline__ unsigned short f2bf(float x) {
  unsigned int b = __builtin_bit_cast(unsigned int, x);
  b += 0x7FFFu + ((b >> 16) & 1u);             // round-to-nearest-even
  return (unsigned short)(b >> 16);
}
__device__ __forceinline__ unsigned int pack2bf(float lo, float hi) {
  return (unsigned int)f2bf(lo) | ((unsigned int)f2bf(hi) << 16);
}

#if __has_builtin(__builtin_amdgcn_global_load_async_to_lds_b128)
#define ASYNC_LDS 1
// Builtin signature: (v4i AS1* src, v4i AS3* dst, imm offset, imm cpol).
// Build AS1/AS3 pointers via integer casts (LDS offset lives in the low 32
// bits of the flat address of a __shared__ object).
typedef __attribute__((address_space(1))) v4i* gv4i_p;
typedef __attribute__((address_space(3))) v4i* lv4i_p;
__device__ __forceinline__ void async_b128(const void* g, const void* l) {
  __builtin_amdgcn_global_load_async_to_lds_b128(
      (gv4i_p)(unsigned long long)g,
      (lv4i_p)(unsigned int)(unsigned long long)l,
      0, 0);
}
#endif

// ---------------------------------------------------------------------------
// C[M,N] = A[M,K](f32) * W[K,N](f32) + bias[N], bf16 WMMA, f32 accumulate.
// Block = 128 threads (4 waves). Block tile 128x64, K-step 32.
// Wave w computes rows w*32..w*32+31 (two 16-row A frags) x full 64 cols.
// 8 WMMA per K-step per wave; each B fragment feeds two WMMAs.
// ---------------------------------------------------------------------------
__global__ __launch_bounds__(128) void gemm_bias_bf16(
    const float* __restrict__ A, const float* __restrict__ W,
    const float* __restrict__ bias, float* __restrict__ C,
    int M, int N, int K)
{
  __shared__ __align__(16) unsigned short As[128][36];  // [m][k], pad 4
  __shared__ __align__(16) unsigned short Bs[64][36];   // transposed: [n][k], pad 4

  const int lane = threadIdx.x & 31;
  const int wv   = threadIdx.x >> 5;
  const int lm   = lane & 15;
  const int hh   = lane >> 4;
  const int mBase = blockIdx.y * 128;
  const int nBase = blockIdx.x * 64;

  f32x8 acc0[4] = {};
  f32x8 acc1[4] = {};

  for (int k0 = 0; k0 < K; k0 += 32) {
    __syncthreads();
#pragma unroll
    for (int i = 0; i < 8; ++i) {               // A tile: 128 x 32 f32 -> bf16
      int idx = (int)threadIdx.x + i * 128;     // 0..1023
      int r  = idx >> 3;                        // 0..127
      int c4 = idx & 7;                         // float4 slot
      const float4 fa = *(const float4*)(A + (size_t)(mBase + r) * K + k0 + c4 * 4);
      *(unsigned int*)&As[r][c4 * 4]     = pack2bf(fa.x, fa.y);
      *(unsigned int*)&As[r][c4 * 4 + 2] = pack2bf(fa.z, fa.w);
    }
#pragma unroll
    for (int i = 0; i < 4; ++i) {               // W tile: 32(k) x 64(n), transposed
      int idx = (int)threadIdx.x + i * 128;     // 0..511
      int rB  = idx >> 4;                       // 0..31 (k)
      int cB4 = idx & 15;                       // 0..15
      const float4 fb = *(const float4*)(W + (size_t)(k0 + rB) * N + nBase + cB4 * 4);
      Bs[cB4 * 4 + 0][rB] = f2bf(fb.x);
      Bs[cB4 * 4 + 1][rB] = f2bf(fb.y);
      Bs[cB4 * 4 + 2][rB] = f2bf(fb.z);
      Bs[cB4 * 4 + 3][rB] = f2bf(fb.w);
    }
    __syncthreads();

    // A fragments (16x32): lanes 0-15 K=0..7,16..23 ; lanes 16-31 K=8..15,24..31
    ABFrag a0, a1;
    const int r0 = wv * 32 + lm;
#pragma unroll
    for (int i = 0; i < 8; ++i) {
      int kk = (i < 4 ? 2 * i : 16 + 2 * (i - 4)) + hh * 8;
      a0.u[i] = *(const unsigned int*)&As[r0][kk];
      a1.u[i] = *(const unsigned int*)&As[r0 + 16][kk];
    }
#pragma unroll
    for (int nt = 0; nt < 4; ++nt) {
      // B fragment (32x16): lane col n = lm; lanes 0-15 K=0..7,16..23 ; lanes 16-31 K=8..15,24..31
      ABFrag b;
      const int col = nt * 16 + lm;
#pragma unroll
      for (int i = 0; i < 8; ++i)
        b.u[i] = *(const unsigned int*)&Bs[col][(i < 4 ? 2 * i : 16 + 2 * (i - 4)) + hh * 8];
      acc0[nt] = __builtin_amdgcn_wmma_f32_16x16x32_bf16(
          false, a0.v, false, b.v, (short)0, acc0[nt], false, false);
      acc1[nt] = __builtin_amdgcn_wmma_f32_16x16x32_bf16(
          false, a1.v, false, b.v, (short)0, acc1[nt], false, false);
    }
  }

#pragma unroll
  for (int nt = 0; nt < 4; ++nt) {
#pragma unroll
    for (int i = 0; i < 8; ++i) {
      int row = mBase + wv * 32 + (hh ? i + 8 : i);
      int col = nBase + nt * 16 + lm;
      float bsv = bias[col];
      C[(size_t)row * N + col]        = acc0[nt][i] + bsv;
      C[(size_t)(row + 16) * N + col] = acc1[nt][i] + bsv;
    }
  }
}

// ---------------------------------------------------------------------------
// RoPE on Q/K projections + repack Q,K,V to bf16 head-major [B,H,S,HD].
// One thread per (b,h,s,j), j in [0,64): handles dims j and j+64.
// ---------------------------------------------------------------------------
__global__ __launch_bounds__(256) void rope_pack(
    const float* __restrict__ Qp, const float* __restrict__ Kp,
    const float* __restrict__ Vp,
    unsigned short* __restrict__ qh, unsigned short* __restrict__ kh,
    unsigned short* __restrict__ vh)
{
  int idx = blockIdx.x * 256 + (int)threadIdx.x;
  int j  = idx & 63;
  int s  = (idx >> 6) & (S_ - 1);
  int bh = idx >> 17;                       // b*16+h
  int h  = bh & (H_ - 1);
  int b  = bh >> 4;
  size_t src = (((size_t)b * S_ + s) * H_ + h) * HD_;   // [B,S,H,HD]
  size_t dst = ((size_t)bh * S_ + s) * HD_;             // [B,H,S,HD]
  float ang = (float)s * __powf(10000.f, -(float)(2 * j) * (1.0f / (float)HD_));
  float sn, cs;
  __sincosf(ang, &sn, &cs);
  float x1 = Qp[src + j], x2 = Qp[src + j + 64];
  qh[dst + j]      = f2bf(x1 * cs - x2 * sn);
  qh[dst + j + 64] = f2bf(x2 * cs + x1 * sn);
  x1 = Kp[src + j]; x2 = Kp[src + j + 64];
  kh[dst + j]      = f2bf(x1 * cs - x2 * sn);
  kh[dst + j + 64] = f2bf(x2 * cs + x1 * sn);
  vh[dst + j]      = f2bf(Vp[src + j]);
  vh[dst + j + 64] = f2bf(Vp[src + j + 64]);
}

// ---------------------------------------------------------------------------
// Flash attention, causal, per (b,h). Block = 128 threads (4 waves), each wave
// owns 16 query rows (64/block). Keys processed 32 at a time through LDS,
// staged with async global->LDS copies when available.
// out written fp32 as [B,S,H*HD] ready for the output projection GEMM.
// ---------------------------------------------------------------------------
__global__ __launch_bounds__(128) void flash_attn(
    const unsigned short* __restrict__ qh, const unsigned short* __restrict__ kh,
    const unsigned short* __restrict__ vh, float* __restrict__ out)
{
  __shared__ __align__(16) unsigned short Ks[32][HD_ + 8];  // [key][d]
  __shared__ __align__(16) unsigned short Vs[32][HD_ + 8];  // [key][d]
  __shared__ __align__(16) unsigned short Ps[4][16][34];    // per-wave P tile [m][kk]

  const int lane = threadIdx.x & 31;
  const int wv   = threadIdx.x >> 5;
  const int lm   = lane & 15;
  const int hh   = lane >> 4;
  const int bh   = blockIdx.y;
  const int q0   = blockIdx.x * 64 + wv * 16;

  const unsigned short* qb = qh + (size_t)bh * S_ * HD_;
  const unsigned short* kb = kh + (size_t)bh * S_ * HD_;
  const unsigned short* vb = vh + (size_t)bh * S_ * HD_;

  // Q fragments: 4 x (16x32) covering HD=128
  ABFrag qf[4];
#pragma unroll
  for (int f = 0; f < 4; ++f)
#pragma unroll
    for (int i = 0; i < 8; ++i) {
      int k = f * 32 + (i < 4 ? 2 * i : 16 + 2 * (i - 4)) + hh * 8;
      qf[f].u[i] = *(const unsigned int*)(qb + (size_t)(q0 + lm) * HD_ + k);
    }

  float rm[8], rl[8];
#pragma unroll
  for (int i = 0; i < 8; ++i) { rm[i] = -3.0e38f; rl[i] = 0.f; }
  f32x8 oacc[8] = {};

  const float scale = 0.08838834764831845f;   // 1/sqrt(128)
  const int kend = (blockIdx.x + 1) * 64;

  for (int kb0 = 0; kb0 < kend; kb0 += 32) {
    __syncthreads();
#if defined(ASYNC_LDS)
#pragma unroll
    for (int i = 0; i < 4; ++i) {
      int idx = (int)threadIdx.x + i * 128;   // 0..511
      int r  = idx >> 4;                      // key row 0..31
      int c8 = idx & 15;                      // 8-elem group
      async_b128(kb + (size_t)(kb0 + r) * HD_ + c8 * 8, &Ks[r][c8 * 8]);
      async_b128(vb + (size_t)(kb0 + r) * HD_ + c8 * 8, &Vs[r][c8 * 8]);
    }
    asm volatile("s_wait_asynccnt 0x0" ::: "memory");
#else
#pragma unroll
    for (int i = 0; i < 4; ++i) {
      int idx = (int)threadIdx.x + i * 128;   // 0..511
      int r  = idx >> 4;                      // key row 0..31
      int c8 = idx & 15;                      // 8-elem group
      *(uint4*)&Ks[r][c8 * 8] = *(const uint4*)(kb + (size_t)(kb0 + r) * HD_ + c8 * 8);
      *(uint4*)&Vs[r][c8 * 8] = *(const uint4*)(vb + (size_t)(kb0 + r) * HD_ + c8 * 8);
    }
#endif
    __syncthreads();

    // scores: two 16x16 subtiles (keys 0-15, 16-31), K=HD via 4 WMMA each
    f32x8 sc[2] = {};
#pragma unroll
    for (int t = 0; t < 2; ++t)
#pragma unroll
      for (int f = 0; f < 4; ++f) {
        ABFrag kf;   // B = K^T (d x key): lane col = key, K-dim = d
#pragma unroll
        for (int i = 0; i < 8; ++i) {
          int d = f * 32 + (i < 4 ? 2 * i : 16 + 2 * (i - 4)) + hh * 8;
          kf.u[i] = *(const unsigned int*)&Ks[t * 16 + lm][d];
        }
        sc[t] = __builtin_amdgcn_wmma_f32_16x16x32_bf16(
            false, qf[f].v, false, kf.v, (short)0, sc[t], false, false);
      }

    // online softmax per row (rows striped: vgpr i -> row i or i+8 per lane half)
#pragma unroll
    for (int i = 0; i < 8; ++i) {
      int m = (hh ? i + 8 : i);
      int qrow = q0 + m;
      float v0 = sc[0][i] * scale;
      float v1 = sc[1][i] * scale;
      if (kb0 + lm > qrow)      v0 = -3.0e38f;
      if (kb0 + 16 + lm > qrow) v1 = -3.0e38f;
      float mx = fmaxf(v0, v1);
#pragma unroll
      for (int off = 1; off < 16; off <<= 1)
        mx = fmaxf(mx, __shfl_xor(mx, off, 32));
      float nm = fmaxf(rm[i], mx);
      float alpha = __expf(rm[i] - nm);
      float p0 = __expf(v0 - nm);
      float p1 = __expf(v1 - nm);
      float sm = p0 + p1;
#pragma unroll
      for (int off = 1; off < 16; off <<= 1)
        sm += __shfl_xor(sm, off, 32);
      rl[i] = rl[i] * alpha + sm;
      rm[i] = nm;
#pragma unroll
      for (int dt = 0; dt < 8; ++dt) oacc[dt][i] = oacc[dt][i] * alpha;
      Ps[wv][m][lm]      = f2bf(p0);      // C-layout -> A-layout via LDS
      Ps[wv][m][16 + lm] = f2bf(p1);
    }

    asm volatile("s_wait_dscnt 0x0" ::: "memory");   // same-wave P RAW guard

    // P as A fragment (16 x 32 keys)
    ABFrag pa;
#pragma unroll
    for (int i = 0; i < 8; ++i) {
      int k = (i < 4 ? 2 * i : 16 + 2 * (i - 4)) + hh * 8;
      pa.u[i] = *(const unsigned int*)&Ps[wv][lm][k];
    }
    // O += P * V : 8 d-tiles, K = 32 keys
#pragma unroll
    for (int dt = 0; dt < 8; ++dt) {
      ABFrag vf;   // B = V (key x d): lane col = d, K-dim = key
      int d = dt * 16 + lm;
#pragma unroll
      for (int i = 0; i < 8; ++i) {
        int k = (i < 4 ? 2 * i : 16 + 2 * (i - 4)) + hh * 8;
        unsigned int lo = Vs[k][d];
        unsigned int hi = Vs[k + 1][d];
        vf.u[i] = lo | (hi << 16);
      }
      oacc[dt] = __builtin_amdgcn_wmma_f32_16x16x32_bf16(
          false, pa.v, false, vf.v, (short)0, oacc[dt], false, false);
    }
  }

  // normalize + write [B,S,H*HD] fp32
  const int b = bh >> 4;
  const int h = bh & (H_ - 1);
#pragma unroll
  for (int i = 0; i < 8; ++i) {
    float inv = 1.0f / rl[i];
    int row = q0 + (hh ? i + 8 : i);
    size_t base = ((size_t)b * S_ + row) * D_ + (size_t)h * HD_;
#pragma unroll
    for (int dt = 0; dt < 8; ++dt)
      out[base + dt * 16 + lm] = oacc[dt][i] * inv;
  }
}

// ---------------------------------------------------------------------------
extern "C" void kernel_launch(void* const* d_in, const int* in_sizes, int n_in,
                              void* d_out, int out_size, void* d_ws, size_t ws_size,
                              hipStream_t stream) {
  (void)in_sizes; (void)n_in; (void)out_size; (void)ws_size;
  const float* queries = (const float*)d_in[0];
  const float* keys    = (const float*)d_in[1];
  const float* values  = (const float*)d_in[2];
  const float* Wq = (const float*)d_in[3];
  const float* bq = (const float*)d_in[4];
  const float* Wk = (const float*)d_in[5];
  const float* bk = (const float*)d_in[6];
  const float* Wv = (const float*)d_in[7];
  const float* bv = (const float*)d_in[8];
  const float* Wo = (const float*)d_in[9];
  const float* bo = (const float*)d_in[10];

  const int M = B_ * S_;                       // 4096
  const size_t tok = (size_t)M * D_;           // 8.39M elements
  float* Qp = (float*)d_ws;
  float* Kp = Qp + tok;
  float* Vp = Kp + tok;
  unsigned short* qh = (unsigned short*)(Vp + tok);
  unsigned short* kh = qh + tok;
  unsigned short* vh = kh + tok;
  float* attn = Qp;                            // reuse: Qp consumed by rope_pack

  dim3 gg(D_ / 64, M / 128);                   // (32, 32)
  gemm_bias_bf16<<<gg, 128, 0, stream>>>(queries, Wq, bq, Qp, M, D_, D_);
  gemm_bias_bf16<<<gg, 128, 0, stream>>>(keys,    Wk, bk, Kp, M, D_, D_);
  gemm_bias_bf16<<<gg, 128, 0, stream>>>(values,  Wv, bv, Vp, M, D_, D_);

  const int total = B_ * H_ * S_ * (HD_ / 2);  // 4,194,304
  rope_pack<<<total / 256, 256, 0, stream>>>(Qp, Kp, Vp, qh, kh, vh);

  flash_attn<<<dim3(S_ / 64, B_ * H_), 128, 0, stream>>>(qh, kh, vh, attn);

  gemm_bias_bf16<<<gg, 128, 0, stream>>>(attn, Wo, bo, (float*)d_out, M, D_, D_);
}
